// SimpleBilateralFilter_38998303048147
// MI455X (gfx1250) — matrix operand order repaired and checked by reading, not compiled, not run on hardware
//
#include <hip/hip_runtime.h>

typedef __attribute__((ext_vector_type(16))) _Float16 v16h;
typedef __attribute__((ext_vector_type(8)))  float    v8f;

#define RAD   8
#define NCH   12
#define TW    16            // output tile width (pixels per wave row)
#define THT   8             // output tile height (one row per wave)
#define GW    32            // guide cols in tile
#define GH    24            // guide rows in tile
#define NPIX  (GH * GW)     // 768
#define BSTR  8             // dwords per pixel in packed-B tile (12 halves + pad)

__global__ __launch_bounds__(256, 1) void bilateral_wmma_kernel(
    const float* __restrict__ in, const float* __restrict__ params,
    float* __restrict__ out, int H, int W)
{
  // WMMA-ready packed guide: dword k of pixel holds halves (K=2k, K=2k+1); [6..7]=0
  __shared__ __align__(32) unsigned int sB[GH][GW][BSTR];   // 24.0 KB
  __shared__ float4 sQ[GH][GW];                             // {Sb, g0, g1, g2} 12 KB
  __shared__ __align__(32) unsigned int sZero[BSTR];        // B source for hi lanes

  const int tid   = threadIdx.x;
  const int lane  = tid & 31;
  const int wv    = tid >> 5;         // wave id = output row within tile
  const int m     = lane & 15;        // own pixel index == own q-column index
  const bool hi   = lane >= 16;
  const int frame = blockIdx.z;
  const int x0    = blockIdx.x * TW;
  const int y0    = blockIdx.y * THT;

  // params pre-scaled by log2(e): everything lives in base-2 log domain
  const float LOG2E = 1.4426950408889634f;
  float wcs[NCH];
#pragma unroll
  for (int c = 0; c < NCH; ++c) wcs[c] = params[c] * LOG2E;
  const float sy2 = params[NCH] * LOG2E;
  const float sx2 = params[NCH + 1] * LOG2E;

  const float* inF = in + (size_t)frame * NCH * H * W;

  if (tid < BSTR) sZero[tid] = 0u;

  // ---- setup pass: build packed B tile + {Sb,colors} tile straight from global ----
#pragma unroll 1
  for (int idx = tid; idx < NPIX; idx += 256) {
    const int rr = idx / GW;
    const int cc = idx % GW;
    const int gy = y0 - RAD + rr;
    const int gx = x0 - RAD + cc;
    const bool inimg = (gy >= 0 && gy < H && gx >= 0 && gx < W);
    float g[NCH];
#pragma unroll
    for (int c = 0; c < NCH; ++c) g[c] = 0.0f;
    if (inimg) {
#pragma unroll
      for (int c = 0; c < NCH; ++c) g[c] = inF[((size_t)c * H + gy) * W + gx];
    }
    float sb;
    if (inimg) {
      sb = 0.0f;
#pragma unroll
      for (int c = 0; c < NCH; ++c) sb = fmaf(wcs[c] * g[c], g[c], sb);
    } else {
      sb = -__builtin_inff();        // exp2(-inf)=0 kills padded taps exactly
    }
#pragma unroll
    for (int k = 0; k < 6; ++k) {
      const auto p = __builtin_amdgcn_cvt_pkrtz(g[2 * k], g[2 * k + 1]);
      sB[rr][cc][k] = __builtin_bit_cast(unsigned int, p);
    }
    sB[rr][cc][6] = 0u;
    sB[rr][cc][7] = 0u;
    sQ[rr][cc] = make_float4(sb, g[0], g[1], g[2]);
  }
  __syncthreads();

  // ---- per-wave p-side data (always in-image; 64B-coalesced, L2 hot) ----
  const int px = x0 + m, py = y0 + wv;
  float gp[NCH];
#pragma unroll
  for (int c = 0; c < NCH; ++c) gp[c] = inF[((size_t)c * H + py) * W + px];

  float sa = 0.0f;
#pragma unroll
  for (int c = 0; c < NCH; ++c) sa = fmaf(wcs[c] * gp[c], gp[c], sa);

  // A matrix (16x32 f16): u_c = (log2e*w_c)*g_c(p); lanes0-15 K=0..11, lanes16-31 zero
  v16h A;
#pragma unroll
  for (int h = 0; h < 16; ++h) A[h] = (_Float16)0.0f;
#pragma unroll
  for (int h = 0; h < 12; ++h) {
    const float u = hi ? 0.0f
                       : wcs[h] * gp[h];
    A[h] = (_Float16)u;
  }

  // pre[half][r] = Sa(row mr) + (|dx|<=8 ? sx2*dx^2 : -inf)  (dy-independent)
  float pre[2][8];
#pragma unroll
  for (int half = 0; half < 2; ++half) {
#pragma unroll
    for (int r = 0; r < 8; ++r) {
      const int mr = r + (hi ? 8 : 0);
      const float saR = __shfl(sa, mr, 32);
      const int dx = half * 16 + m - mr - RAD;
      const float sp = (dx >= -RAD && dx <= RAD) ? sx2 * (float)(dx * dx)
                                                 : -__builtin_inff();
      pre[half][r] = saR + sp;
    }
  }

  float accD[2][8], acc0[2][8], acc1[2][8], acc2[2][8];
#pragma unroll
  for (int half = 0; half < 2; ++half)
#pragma unroll
    for (int r = 0; r < 8; ++r) {
      accD[half][r] = 0.0f; acc0[half][r] = 0.0f;
      acc1[half][r] = 0.0f; acc2[half][r] = 0.0f;
    }

  auto ldB = [&](int tr, int col) -> v16h {
    const unsigned int* p = hi ? sZero : &sB[tr][col][0];
    return *(const v16h*)p;
  };

  // ---- dy loop, software-pipelined (next row's B/Q loads overlap exp block) ----
  const int prow = wv + RAD;
  int tr = prow - RAD;
  v16h   B0 = ldB(tr, m), B1 = ldB(tr, 16 + m);
  float4 q0 = sQ[tr][m],  q1 = sQ[tr][16 + m];

#pragma unroll 1
  for (int dy = -RAD; dy <= RAD; ++dy) {
    const int trn = (dy < RAD) ? (tr + 1) : tr;
    const v16h   nB0 = ldB(trn, m), nB1 = ldB(trn, 16 + m);
    const float4 nq0 = sQ[trn][m],  nq1 = sQ[trn][16 + m];

    v8f Z;
#pragma unroll
    for (int r = 0; r < 8; ++r) Z[r] = 0.0f;
    const v8f C0 = __builtin_amdgcn_wmma_f32_16x16x32_f16(
        false, A, false, B0, (short)0, Z, false, false);
    const v8f C1 = __builtin_amdgcn_wmma_f32_16x16x32_f16(
        false, A, false, B1, (short)0, Z, false, false);

    const float sdy = sy2 * (float)(dy * dy);
    const float t0 = q0.x + sdy;
    const float t1 = q1.x + sdy;

#pragma unroll
    for (int r = 0; r < 8; ++r) {
      const float w0 = __builtin_amdgcn_exp2f(fmaf(-2.0f, C0[r], pre[0][r] + t0));
      accD[0][r] += w0;
      acc0[0][r] = fmaf(w0, q0.y, acc0[0][r]);
      acc1[0][r] = fmaf(w0, q0.z, acc1[0][r]);
      acc2[0][r] = fmaf(w0, q0.w, acc2[0][r]);
      const float w1 = __builtin_amdgcn_exp2f(fmaf(-2.0f, C1[r], pre[1][r] + t1));
      accD[1][r] += w1;
      acc0[1][r] = fmaf(w1, q1.y, acc0[1][r]);
      acc1[1][r] = fmaf(w1, q1.z, acc1[1][r]);
      acc2[1][r] = fmaf(w1, q1.w, acc2[1][r]);
    }

    B0 = nB0; B1 = nB1; q0 = nq0; q1 = nq1;
    tr = trn;
  }

  // ---- reduce over the 16 q-columns per 16-lane half, write out ----
  const size_t HW = (size_t)H * W;
#pragma unroll
  for (int r = 0; r < 8; ++r) {
    float d  = accD[0][r] + accD[1][r];
    float n0 = acc0[0][r] + acc0[1][r];
    float n1 = acc1[0][r] + acc1[1][r];
    float n2 = acc2[0][r] + acc2[1][r];
#pragma unroll
    for (int s = 8; s >= 1; s >>= 1) {   // xor masks stay within 16-lane halves
      d  += __shfl_xor(d,  s, 32);
      n0 += __shfl_xor(n0, s, 32);
      n1 += __shfl_xor(n1, s, 32);
      n2 += __shfl_xor(n2, s, 32);
    }
    const int mr = r + (hi ? 8 : 0);
    if (m == mr) {                       // owner lane has albedo (gp[3..5]) in regs
      const float inv = 1.0f / d;
      const size_t base = ((size_t)(frame * 3) * H + (y0 + wv)) * W + (x0 + mr);
      out[base]        = gp[3] * n0 * inv;
      out[base + HW]   = gp[4] * n1 * inv;
      out[base + 2*HW] = gp[5] * n2 * inv;
    }
  }
}

extern "C" void kernel_launch(void* const* d_in, const int* in_sizes, int n_in,
                              void* d_out, int out_size, void* d_ws, size_t ws_size,
                              hipStream_t stream) {
  const float* in     = (const float*)d_in[0];   // (1,24,720,1280) f32
  const float* params = (const float*)d_in[1];   // (28,) f32; only [:14] used
  float* out          = (float*)d_out;           // (1,6,720,1280) f32
  (void)in_sizes; (void)n_in; (void)out_size; (void)d_ws; (void)ws_size;

  const int H = 720, W = 1280;
  dim3 grid(W / TW, H / THT, 2);
  bilateral_wmma_kernel<<<grid, dim3(256, 1, 1), 0, stream>>>(in, params, out, H, W);
}
